// CharRNN_30992484008265
// MI455X (gfx1250) — matrix operand order repaired and checked by reading, not compiled, not run on hardware
//
#include <hip/hip_runtime.h>

// ---------------- problem constants ----------------
#define Hc   1024            // hidden
#define Bc   32              // batch
#define Tc   128             // time steps
#define Vc   32000           // vocab
#define Lc   2               // lstm layers
#define K2   2048            // 2*H  (concat input width)
#define G4   4096            // 4*H  (gates width)
#define LPAD 8               // LDS row padding (bf16 elems)

typedef __attribute__((ext_vector_type(16))) __bf16        v16bf;
typedef __attribute__((ext_vector_type(8)))  float         v8f;
typedef __attribute__((ext_vector_type(4)))  unsigned int  u32x4;

struct V16pair { u32x4 q0, q1; };

// float -> bf16, round-to-nearest-even
static __device__ __forceinline__ __bf16 f2bf(float f) {
    unsigned u = __builtin_bit_cast(unsigned, f);
    unsigned r = u + 0x7FFFu + ((u >> 16) & 1u);
    unsigned short s = (unsigned short)(r >> 16);
    return __builtin_bit_cast(__bf16, s);
}

static __device__ __forceinline__ float sigm(float x) {
    return 1.0f / (1.0f + __expf(-x));
}

// A fragment (16x32 bf16, MxK): lane = row M (lanes 16-31 mirror rows 0-15).
// lo lanes hold K 0..15, hi lanes K 16..31 (matches B fragment)
// => two 16B chunks at element offsets k0+16*hi and k0+16*hi+8.
static __device__ __forceinline__ v16bf load_a_frag(const __bf16* rowptr, int k0, int hi) {
    V16pair p;
    p.q0 = *(const u32x4*)(rowptr + k0 + hi * 16);
    p.q1 = *(const u32x4*)(rowptr + k0 + hi * 16 + 8);
    return __builtin_bit_cast(v16bf, p);
}

// B fragment (32x16 bf16, KxN), weights stored transposed [N][K]:
// lane = column N; lo lanes hold K 0..15, hi lanes K 16..31 => one
// contiguous 32B run at k0 + 16*hi.
static __device__ __forceinline__ v16bf load_b_frag(const __bf16* rowptr, int k0, int hi) {
    V16pair p;
    const __bf16* q = rowptr + k0 + hi * 16;
    p.q0 = *(const u32x4*)(q);
    p.q1 = *(const u32x4*)(q + 8);
    return __builtin_bit_cast(v16bf, p);
}

#define WMMA_BF16(A, B, C) \
    __builtin_amdgcn_wmma_f32_16x16x32_bf16(false, (A), false, (B), (short)0, (C), false, false)

// ---------------- weight / activation prep ----------------

// lstm_w [L][2H][4H] f32  ->  Wt [L][4H][2H] bf16 (transposed, K-contiguous)
__global__ __launch_bounds__(256) void conv_lstm_w_kernel(const float* __restrict__ lw,
                                                          __bf16* __restrict__ Wt) {
    size_t idx = (size_t)blockIdx.x * blockDim.x + threadIdx.x;
    size_t total = (size_t)Lc * G4 * K2;
    if (idx >= total) return;
    int k = (int)(idx % K2);
    size_t r = idx / K2;
    int n = (int)(r % G4);
    int l = (int)(r / G4);
    Wt[idx] = f2bf(lw[((size_t)l * K2 + k) * G4 + n]);
}

// softmax_w [V][H] f32 -> bf16 (already N-major, K-contiguous for B operand)
__global__ __launch_bounds__(256) void conv_soft_w_kernel(const float* __restrict__ sw,
                                                          __bf16* __restrict__ Ws) {
    size_t idx = (size_t)blockIdx.x * blockDim.x + threadIdx.x;
    if (idx >= (size_t)Vc * Hc) return;
    Ws[idx] = f2bf(sw[idx]);
}

// embedding gather: ids [B][T] -> xb [B][T][H] bf16
__global__ __launch_bounds__(256) void embed_kernel(const int* __restrict__ ids,
                                                    const float* __restrict__ emb,
                                                    __bf16* __restrict__ xb) {
    size_t idx = (size_t)blockIdx.x * blockDim.x + threadIdx.x;
    if (idx >= (size_t)Bc * Tc * Hc) return;
    int tok = ids[idx / Hc];
    xb[idx] = f2bf(emb[(size_t)tok * Hc + (idx % Hc)]);
}

// zero c-state (f32) and parity-0 h buffers (bf16)
__global__ __launch_bounds__(256) void init_state_kernel(float* __restrict__ c,
                                                         __bf16* __restrict__ h) {
    int i = blockIdx.x * blockDim.x + threadIdx.x;
    const int n = Lc * Bc * Hc;
    if (i < n) {
        c[i] = 0.0f;
        int l = i / (Bc * Hc), r = i % (Bc * Hc);
        h[(size_t)(l * 2 + 0) * (Bc * Hc) + r] = f2bf(0.0f);
    }
}

// ---------------- fused LSTM cell step ----------------
// gates[B][4H] = [x_t | h_prev] @ W^T + b ; each wave owns one 16x16 output
// column tile n0 and computes it for ALL FOUR gates (4 accumulators, one
// shared A fragment), then performs the cell update in registers.
__global__ __launch_bounds__(128) void lstm_step_kernel(
    const __bf16* __restrict__ inp, long long inp_stride,   // input half of concat
    const __bf16* __restrict__ hprev,                       // [B][H] bf16
    const __bf16* __restrict__ Wt,                          // [4H][2H] bf16
    const float*  __restrict__ bias,                        // [4H]
    float*        __restrict__ cstate,                      // [B][H] f32 (in place)
    __bf16*       __restrict__ hnext,                       // [B][H] bf16
    __bf16*       __restrict__ act_out,                     // [B][T][H] bf16 or null
    int t)
{
    int wid    = blockIdx.x * (blockDim.x >> 5) + (threadIdx.x >> 5);
    int lane   = threadIdx.x & 31;
    int lane15 = lane & 15;
    int hi     = lane >> 4;
    int ntile  = wid & 63;   // 64 column tiles over H
    int mtile  = wid >> 6;   // 2 row tiles over B=32
    int m0 = mtile * 16;
    int n0 = ntile * 16;

    const __bf16* arow_x = inp   + (long long)(m0 + lane15) * inp_stride;
    const __bf16* arow_h = hprev + (size_t)(m0 + lane15) * Hc;
    const __bf16* brow0 = Wt + (size_t)(0 * Hc + n0 + lane15) * K2;
    const __bf16* brow1 = Wt + (size_t)(1 * Hc + n0 + lane15) * K2;
    const __bf16* brow2 = Wt + (size_t)(2 * Hc + n0 + lane15) * K2;
    const __bf16* brow3 = Wt + (size_t)(3 * Hc + n0 + lane15) * K2;

    v8f acc0 = {}, acc1 = {}, acc2 = {}, acc3 = {};

    // K-loop split into two straight-line halves (no per-iteration select):
    // first the x_t half (K = 0..H), then the h_prev half (K = H..2H).
#pragma unroll 2
    for (int k0 = 0; k0 < Hc; k0 += 32) {
        v16bf a  = load_a_frag(arow_x, k0, hi);
        v16bf b0 = load_b_frag(brow0, k0, hi);
        v16bf b1 = load_b_frag(brow1, k0, hi);
        v16bf b2 = load_b_frag(brow2, k0, hi);
        v16bf b3 = load_b_frag(brow3, k0, hi);
        acc0 = WMMA_BF16(a, b0, acc0);
        acc1 = WMMA_BF16(a, b1, acc1);
        acc2 = WMMA_BF16(a, b2, acc2);
        acc3 = WMMA_BF16(a, b3, acc3);
    }
#pragma unroll 2
    for (int k0 = 0; k0 < Hc; k0 += 32) {
        v16bf a  = load_a_frag(arow_h, k0, hi);
        v16bf b0 = load_b_frag(brow0, Hc + k0, hi);
        v16bf b1 = load_b_frag(brow1, Hc + k0, hi);
        v16bf b2 = load_b_frag(brow2, Hc + k0, hi);
        v16bf b3 = load_b_frag(brow3, Hc + k0, hi);
        acc0 = WMMA_BF16(a, b0, acc0);
        acc1 = WMMA_BF16(a, b1, acc1);
        acc2 = WMMA_BF16(a, b2, acc2);
        acc3 = WMMA_BF16(a, b3, acc3);
    }

    int n = n0 + lane15;
    float bi = bias[0 * Hc + n];
    float bj = bias[1 * Hc + n];
    float bf_ = bias[2 * Hc + n];
    float bo = bias[3 * Hc + n];

    // C/D layout: VGPR r -> M = m0 + r + 8*hi, N = n0 + lane15
#pragma unroll
    for (int r = 0; r < 8; ++r) {
        int m = m0 + hi * 8 + r;
        float gi = acc0[r] + bi;
        float gj = acc1[r] + bj;
        float gf = acc2[r] + bf_;
        float go = acc3[r] + bo;
        size_t idx = (size_t)m * Hc + n;
        float cold = cstate[idx];
        float cnew = sigm(gf + 1.0f) * cold + sigm(gi) * tanhf(gj);
        cstate[idx] = cnew;
        float hv = sigm(go) * tanhf(cnew);
        __bf16 hb = f2bf(hv);
        hnext[idx] = hb;
        if (act_out) act_out[((size_t)m * Tc + t) * Hc + n] = hb;  // row b*T+t
    }
}

// ---------------- logits GEMM ----------------
// out[4096][32000] = act[4096][1024] @ Ws^T + bias.
// Block: 256 thr (8 waves), one 16-row A tile staged in LDS via the gfx1250
// async copy path (global_load_async_to_lds_b128, ASYNCcnt), each wave owns
// 2 column tiles (16 tiles / block => 256 cols). Grid (32000/256, 4096/16).
__global__ __launch_bounds__(256) void logits_kernel(const __bf16* __restrict__ act,
                                                     const __bf16* __restrict__ Ws,
                                                     const float*  __restrict__ sbias,
                                                     float*        __restrict__ out) {
    __shared__ __align__(16) __bf16 As[16][Hc + LPAD];

    int m0 = blockIdx.y * 16;

    // async A-tile staging: 16 rows x 1024 bf16 = 32 KB, 8 x b128 per lane,
    // DMA'd straight into LDS (no VGPR round-trip), tracked by ASYNCcnt.
    {
        int tid = threadIdx.x;
        int row = tid >> 4;            // 0..15
        int col = (tid & 15) * 64;     // 64 bf16 (128 B) per thread
        const __bf16* src = act + (size_t)(m0 + row) * Hc + col;
        // generic LDS address: low 32 bits are the LDS byte offset
        unsigned lds_off = (unsigned)(unsigned long long)(&As[row][col]);
#pragma unroll
        for (int i = 0; i < 8; ++i) {
            unsigned dst = lds_off + (unsigned)(i * 16);
            const __bf16* g = src + i * 8;
            asm volatile("global_load_async_to_lds_b128 %0, %1, off"
                         :: "v"(dst), "v"(g) : "memory");
        }
        asm volatile("s_wait_asynccnt 0x0" ::: "memory");
    }
    __syncthreads();

    int wave   = threadIdx.x >> 5;
    int lane   = threadIdx.x & 31;
    int lane15 = lane & 15;
    int hi     = lane >> 4;
    int nt     = blockIdx.x * 16 + wave * 2;
    int n0a    = (nt + 0) * 16;
    int n0b    = (nt + 1) * 16;

    const __bf16* browa = Ws + (size_t)(n0a + lane15) * Hc;
    const __bf16* browb = Ws + (size_t)(n0b + lane15) * Hc;
    const __bf16* arow  = &As[lane15][0];

    v8f acca = {}, accb = {};
#pragma unroll 4
    for (int k0 = 0; k0 < Hc; k0 += 32) {
        if (k0 + 256 < Hc) {           // in-row weight prefetch, clamped
            __builtin_prefetch(browa + k0 + 256, 0, 1);
            __builtin_prefetch(browb + k0 + 256, 0, 1);
        }
        v16bf a  = load_a_frag(arow, k0, hi);   // ds_load from LDS
        v16bf ba = load_b_frag(browa, k0, hi);
        v16bf bb = load_b_frag(browb, k0, hi);
        acca = WMMA_BF16(a, ba, acca);
        accb = WMMA_BF16(a, bb, accb);
    }

    int na = n0a + lane15;
    int nb = n0b + lane15;
    float ba = sbias[na];
    float bb = sbias[nb];
    // streaming 524 MB output: non-temporal stores so Ws stays L2-resident
#pragma unroll
    for (int r = 0; r < 8; ++r) {
        int m = m0 + hi * 8 + r;
        __builtin_nontemporal_store(acca[r] + ba, &out[(size_t)m * Vc + na]);
        __builtin_nontemporal_store(accb[r] + bb, &out[(size_t)m * Vc + nb]);
    }
}

// ---------------- host launcher ----------------
extern "C" void kernel_launch(void* const* d_in, const int* in_sizes, int n_in,
                              void* d_out, int out_size, void* d_ws, size_t ws_size,
                              hipStream_t stream) {
    const int*   ids = (const int*)d_in[0];     // [B][T]
    const float* emb = (const float*)d_in[1];   // [V][H]
    const float* sw  = (const float*)d_in[2];   // [V][H]
    const float* sb  = (const float*)d_in[3];   // [V]
    const float* lw  = (const float*)d_in[4];   // [L][2H][4H]
    const float* lb  = (const float*)d_in[5];   // [L][4H]
    float* out = (float*)d_out;
    (void)in_sizes; (void)n_in; (void)out_size; (void)ws_size;

    char* p = (char*)d_ws;
    auto alloc = [&](size_t bytes) {
        char* r = p;
        p += (bytes + 255) & ~(size_t)255;
        return r;
    };
    __bf16* Wt   = (__bf16*)alloc(sizeof(__bf16) * (size_t)Lc * G4 * K2);   // 33.5 MB
    __bf16* Ws   = (__bf16*)alloc(sizeof(__bf16) * (size_t)Vc * Hc);        // 65.5 MB
    __bf16* xb   = (__bf16*)alloc(sizeof(__bf16) * (size_t)Bc * Tc * Hc);   //  8.4 MB
    __bf16* actb = (__bf16*)alloc(sizeof(__bf16) * (size_t)Bc * Tc * Hc);   //  8.4 MB
    __bf16* hbuf = (__bf16*)alloc(sizeof(__bf16) * (size_t)Lc * 2 * Bc * Hc);
    float*  cbuf = (float*) alloc(sizeof(float)  * (size_t)Lc * Bc * Hc);

    // state init (must run every call: c/h accumulate across the T loop)
    init_state_kernel<<<(Lc * Bc * Hc + 255) / 256, 256, 0, stream>>>(cbuf, hbuf);

    // weight conversion / transpose
    {
        size_t total = (size_t)Lc * G4 * K2;
        conv_lstm_w_kernel<<<(unsigned)((total + 255) / 256), 256, 0, stream>>>(lw, Wt);
    }
    {
        size_t total = (size_t)Vc * Hc;
        conv_soft_w_kernel<<<(unsigned)((total + 255) / 256), 256, 0, stream>>>(sw, Ws);
    }
    // embedding gather -> bf16 activations
    {
        size_t total = (size_t)Bc * Tc * Hc;
        embed_kernel<<<(unsigned)((total + 255) / 256), 256, 0, stream>>>(ids, emb, xb);
    }

    // recurrent sweep: 2 fused cell launches per step, h double-buffered
    const size_t BH = (size_t)Bc * Hc;
    for (int t = 0; t < Tc; ++t) {
        int pr = t & 1;       // read parity
        int pw = pr ^ 1;      // write parity
        // layer 0: input = x_t (rows strided by T*H)
        lstm_step_kernel<<<32, 128, 0, stream>>>(
            xb + (size_t)t * Hc, (long long)Tc * Hc,
            hbuf + (0 * 2 + pr) * BH,
            Wt + (size_t)0 * G4 * K2, lb + 0 * G4,
            cbuf + 0 * BH,
            hbuf + (0 * 2 + pw) * BH,
            (__bf16*)nullptr, t);
        // layer 1: input = layer-0 h just produced
        lstm_step_kernel<<<32, 128, 0, stream>>>(
            hbuf + (0 * 2 + pw) * BH, (long long)Hc,
            hbuf + (1 * 2 + pr) * BH,
            Wt + (size_t)1 * G4 * K2, lb + 1 * G4,
            cbuf + 1 * BH,
            hbuf + (1 * 2 + pw) * BH,
            actb, t);
    }

    // output projection: [4096,1024] @ [1024,32000] + bias
    logits_kernel<<<dim3(Vc / 256, (Bc * Tc) / 16), 256, 0, stream>>>(actb, Ws, sb, out);
}
